// ArabicMamba_16638703305495
// MI455X (gfx1250) — compile-verified, matrix operand order checked by reading, not run on hardware
//
#include <hip/hip_runtime.h>
#include <hip/hip_bf16.h>

// ============================================================================
// ArabicMamba forward for MI455X (gfx1250, wave32, WMMA).
//
// All dense projections (input_proj, conv3-as-GEMM, skip 1x1, in_proj, x_proj,
// dt, out_proj, cls1, cls2) go through v_wmma_f32_16x16x32_f16 (f16 inputs,
// f32 accumulate). Each wave computes a 16x64 output strip: one A fragment is
// reused against 4 B fragments -> 4 WMMAs per K-step. Fragment loads are
// unpredicated global_load_b128 (clamped addresses + store-side masking), so
// the memory pipeline stays full instead of serializing on per-element
// exec-masked loads. Selective scan / depthwise conv / BN / LN are VALU
// kernels (scan is the transcendental-bound part: ~1.88G expf across 16384
// independent channel threads).
// ============================================================================

typedef _Float16 v16h __attribute__((ext_vector_type(16)));
typedef float    v8f  __attribute__((ext_vector_type(8)));
typedef float    v4f  __attribute__((ext_vector_type(4)));

__device__ __forceinline__ float act_apply(float x, int act) {
  if (act == 1) return x > 0.f ? x : 0.f;                      // relu
  if (act == 2) return x / (1.f + __expf(-x));                 // silu
  if (act == 3) return (x > 20.f) ? x : log1pf(__expf(x));     // softplus
  return x;
}

// Pack four float4 runs into the CDNA5 16-bit A/B fragment ordering
// (f[0..7] = K run at kk+kb, f[8..15] = K run at kk+16+kb).  SC applies the
// loop-invariant row mask (conv time-shift taps that fall off the sequence).
template <bool SC>
__device__ __forceinline__ v16h cvt16(v4f a0, v4f a1, v4f a2, v4f a3, float s) {
  v16h f;
#pragma unroll
  for (int j = 0; j < 4; ++j) {
    float x0 = a0[j], x1 = a1[j], x2 = a2[j], x3 = a3[j];
    if (SC) { x0 *= s; x1 *= s; x2 *= s; x3 *= s; }
    f[j]      = (_Float16)x0;
    f[j + 4]  = (_Float16)x1;
    f[j + 8]  = (_Float16)x2;
    f[j + 12] = (_Float16)x3;
  }
  return f;
}

// ---- Generic WMMA GEMM:  C[M,N] = act( (accum?C:0) + A[M,K] @ W[N,K]^T + bias )
// A row stride lda (column views via pointer offset).  W element (n,k) at
// W[n*ldw + k*wks] (wks=3 slices one tap of a k=3 conv).  tshift shifts A's
// row in time within its length-T sequence (conv-as-GEMM).
// KAL: K % 32 == 0 -> aligned float4 fragment loads.  SHIFT: tshift != 0.
template <bool KAL, bool SHIFT>
__global__ void k_gemm(const float* __restrict__ A, long lda,
                       const float* __restrict__ W, long ldw, long wks,
                       const float* __restrict__ bias,
                       float* __restrict__ C, long ldc,
                       int M, int N, int K, int tshift, int T, int act,
                       int accum) {
  const int wave = threadIdx.x >> 5;
  const int lane = threadIdx.x & 31;
  const int tiles_n = (N + 15) >> 4;
  const int ngrp = (tiles_n + 3) >> 2;              // 4 N-tiles per wave
  const long njobs = (long)((M + 15) >> 4) * ngrp;
  long job = (long)blockIdx.x * (blockDim.x >> 5) + wave;
  if (job >= njobs) return;                         // wave-uniform exit
  const int tm = (int)(job / ngrp);
  const int tg = (int)(job % ngrp);
  const int m0 = tm << 4;
  const int n0 = tg << 6;

  // Loop-invariant per-lane A row (clamped; out-of-range rows masked at store)
  int r = m0 + (lane & 15);
  if (r >= M) r = M - 1;
  long rowIdx = r;
  float rowscale = 1.f;
  if (SHIFT) {
    int t = r % T + tshift;
    if (t >= 0 && t < T) rowIdx = (long)(r - (r % T)) + t;
    else                 rowscale = 0.f;            // tap off sequence -> zero
  }
  const float* Ar = A + rowIdx * lda;
  const int kb = (lane >> 4) << 3;

  // Clamped B row pointers for the 4 N-subtiles (garbage cols masked at store)
  const float* Wr[4];
#pragma unroll
  for (int nt = 0; nt < 4; ++nt) {
    int n = n0 + (nt << 4) + (lane & 15);
    if (n >= N) n = N - 1;
    Wr[nt] = W + (long)n * ldw;
  }

  v8f acc[4] = {{}, {}, {}, {}};
  for (int kk = 0; kk < K; kk += 32) {
    v16h a, b[4];
    if (KAL) {
      const float* Ap = Ar + kk + kb;
      if (kk + 32 < K) __builtin_prefetch(Ap + 32, 0, 1);
      a = cvt16<SHIFT>(*(const v4f*)(Ap),      *(const v4f*)(Ap + 4),
                       *(const v4f*)(Ap + 16), *(const v4f*)(Ap + 20),
                       rowscale);
      if (wks == 1) {
#pragma unroll
        for (int nt = 0; nt < 4; ++nt) {
          const float* Wp = Wr[nt] + kk + kb;
          b[nt] = cvt16<false>(*(const v4f*)(Wp),      *(const v4f*)(Wp + 4),
                               *(const v4f*)(Wp + 16), *(const v4f*)(Wp + 20),
                               1.f);
        }
      } else {                                      // conv tap: k-stride = 3
#pragma unroll
        for (int nt = 0; nt < 4; ++nt) {
#pragma unroll
          for (int j = 0; j < 8; ++j) {
            b[nt][j]     = (_Float16)Wr[nt][(long)(kk + kb + j) * wks];
            b[nt][j + 8] = (_Float16)Wr[nt][(long)(kk + 16 + kb + j) * wks];
          }
        }
      }
    } else {
      // K-edge path: unconditional clamped loads + select-to-zero (no exec
      // branching around loads).
#pragma unroll
      for (int j = 0; j < 8; ++j) {
        int k0 = kk + kb + j, k1 = kk + 16 + kb + j;
        int k0c = k0 < K ? k0 : K - 1;
        int k1c = k1 < K ? k1 : K - 1;
        float va0 = Ar[k0c] * (SHIFT ? rowscale : 1.f);
        float va1 = Ar[k1c] * (SHIFT ? rowscale : 1.f);
        a[j]     = (k0 < K) ? (_Float16)va0 : (_Float16)0.f;
        a[j + 8] = (k1 < K) ? (_Float16)va1 : (_Float16)0.f;
#pragma unroll
        for (int nt = 0; nt < 4; ++nt) {
          float vb0 = Wr[nt][(long)k0c * wks];
          float vb1 = Wr[nt][(long)k1c * wks];
          b[nt][j]     = (k0 < K) ? (_Float16)vb0 : (_Float16)0.f;
          b[nt][j + 8] = (k1 < K) ? (_Float16)vb1 : (_Float16)0.f;
        }
      }
    }
    acc[0] = __builtin_amdgcn_wmma_f32_16x16x32_f16(false, a, false, b[0],
                                                    (short)0, acc[0], false, false);
    acc[1] = __builtin_amdgcn_wmma_f32_16x16x32_f16(false, a, false, b[1],
                                                    (short)0, acc[1], false, false);
    acc[2] = __builtin_amdgcn_wmma_f32_16x16x32_f16(false, a, false, b[2],
                                                    (short)0, acc[2], false, false);
    acc[3] = __builtin_amdgcn_wmma_f32_16x16x32_f16(false, a, false, b[3],
                                                    (short)0, acc[3], false, false);
  }

  const int rbase = m0 + ((lane >> 4) << 3);
#pragma unroll
  for (int nt = 0; nt < 4; ++nt) {
    int col = n0 + (nt << 4) + (lane & 15);
    if (col < N) {
      float bv = bias ? bias[col] : 0.f;
#pragma unroll
      for (int i = 0; i < 8; ++i) {
        int row = rbase + i;
        if (row < M) {
          float v = acc[nt][i] + bv;
          if (accum) v += C[(long)row * ldc + col];
          C[(long)row * ldc + col] = act_apply(v, act);
        }
      }
    }
  }
}

// ---- BN statistics: per-channel mean / biased var over all BT rows ----
__global__ void k_bnstats(const float* __restrict__ x, int C, long R,
                          float* __restrict__ mean, float* __restrict__ var) {
  const int c = blockIdx.x;
  float s = 0.f, s2 = 0.f;
  for (long r = threadIdx.x; r < R; r += blockDim.x) {
    float v = x[r * C + c];
    s += v; s2 += v * v;
  }
  __shared__ float sh[256], sh2[256];
  sh[threadIdx.x] = s; sh2[threadIdx.x] = s2;
  __syncthreads();
  for (int o = 128; o > 0; o >>= 1) {
    if ((int)threadIdx.x < o) { sh[threadIdx.x] += sh[threadIdx.x + o];
                                sh2[threadIdx.x] += sh2[threadIdx.x + o]; }
    __syncthreads();
  }
  if (threadIdx.x == 0) {
    float m = sh[0] / (float)R;
    mean[c] = m;
    var[c]  = sh2[0] / (float)R - m * m;
  }
}

__global__ void k_bn_relu_add(const float* __restrict__ conv,
                              const float* __restrict__ mean,
                              const float* __restrict__ var,
                              const float* __restrict__ g,
                              const float* __restrict__ bta,
                              const float* __restrict__ skip,
                              float* __restrict__ out, int C, long total) {
  long idx = (long)blockIdx.x * blockDim.x + threadIdx.x;
  if (idx >= total) return;
  int c = (int)(idx % C);
  float v = (conv[idx] - mean[c]) * rsqrtf(var[c] + 1e-5f) * g[c] + bta[c];
  v = v > 0.f ? v : 0.f;
  out[idx] = v + skip[idx];
}

// ---- Depthwise causal conv k=4 + bias + SiLU. backward=1 computes the
// time-reversed branch directly in natural storage order. ----
__global__ void k_dwconv_silu(const float* __restrict__ xz, int xz_stride,
                              const float* __restrict__ w,
                              const float* __restrict__ bconv,
                              float* __restrict__ out,
                              int T, int di, long total, int backward) {
  long idx = (long)blockIdx.x * blockDim.x + threadIdx.x;
  if (idx >= total) return;
  int d  = (int)(idx % di);
  long r = idx / di;              // b*T + t
  int t  = (int)(r % T);
  long base = r - t;              // b*T
  float acc = bconv[d];
#pragma unroll
  for (int j = 0; j < 4; ++j) {
    int tt = backward ? (t + 3 - j) : (t - 3 + j);
    if (tt >= 0 && tt < T)
      acc += w[d * 4 + j] * xz[(base + tt) * (long)xz_stride + d];
  }
  out[r * (long)di + d] = acc / (1.f + __expf(-acc));   // silu
}

// ---- Selective scan: one thread per (b, d) channel, 16-state recurrence in
// registers.  y overwrites dt in place (read-before-write per element). ----
__global__ void k_scan(const float* __restrict__ u, float* __restrict__ dty,
                       const float* __restrict__ dbl, int dbl_stride, int dr,
                       const float* __restrict__ A_log, const float* __restrict__ Dp,
                       int Bb, int T, int di, int backward) {
  int idx = blockIdx.x * blockDim.x + threadIdx.x;
  if (idx >= Bb * di) return;
  int d = idx % di, b = idx / di;
  float A[16], h[16];
#pragma unroll
  for (int n = 0; n < 16; ++n) { A[n] = -__expf(A_log[d * 16 + n]); h[n] = 0.f; }
  float Dd = Dp[d];
  for (int tt = 0; tt < T; ++tt) {
    int t = backward ? (T - 1 - tt) : tt;
    long r = (long)b * T + t;
    float uv  = u[r * di + d];
    float dtv = dty[r * di + d];
    const float* dblr = dbl + r * (long)dbl_stride;
    float du = dtv * uv;
    float y = 0.f;
#pragma unroll
    for (int n = 0; n < 16; ++n) {
      h[n] = __expf(dtv * A[n]) * h[n] + du * dblr[dr + n];
      y += h[n] * dblr[dr + 16 + n];
    }
    dty[r * di + d] = y + uv * Dd;
  }
}

// ---- (yf + yb) * silu(z) ----
__global__ void k_gate(const float* __restrict__ yf, const float* __restrict__ yb,
                       const float* __restrict__ xz, int xz_stride, int di,
                       float* __restrict__ out, long total) {
  long idx = (long)blockIdx.x * blockDim.x + threadIdx.x;
  if (idx >= total) return;
  int d  = (int)(idx % di);
  long r = idx / di;
  float z = xz[r * (long)xz_stride + di + d];
  out[idx] = (yf[idx] + yb[idx]) * (z / (1.f + __expf(-z)));
}

// ---- LayerNorm over channels + residual add. One wave32 per row, shuffle
// tree reduction (warpSize == 32 on gfx1250). ----
__global__ void k_ln_add(const float* __restrict__ y, const float* __restrict__ g,
                         const float* __restrict__ bta, const float* __restrict__ hb,
                         float* __restrict__ out, int C, long R) {
  int wave = threadIdx.x >> 5, lane = threadIdx.x & 31;
  long r = (long)blockIdx.x * (blockDim.x >> 5) + wave;
  if (r >= R) return;
  const float* yr = y + r * (long)C;
  float s = 0.f, s2 = 0.f;
  for (int c = lane; c < C; c += 32) { float v = yr[c]; s += v; s2 += v * v; }
#pragma unroll
  for (int m = 16; m > 0; m >>= 1) {
    s  += __shfl_xor(s,  m, 32);
    s2 += __shfl_xor(s2, m, 32);
  }
  float mu  = s / (float)C;
  float inv = rsqrtf(s2 / (float)C - mu * mu + 1e-6f);
  for (int c = lane; c < C; c += 32)
    out[r * (long)C + c] = hb[r * (long)C + c] + g[c] * (yr[c] - mu) * inv + bta[c];
}

// ============================================================================
// Host orchestration
// ============================================================================

struct BlockP {
  const float *bn_b, *bn_g, *conv_w, *ln_b, *ln_g, *skip_w;
  const float *A_b_log, *A_log, *D, *D_b, *conv_b, *conv_b_b,
              *dconv_w, *dconv_w_b, *dt_b, *dt_b_b, *dt_w, *dt_w_b,
              *in_proj, *out_proj, *x_proj, *x_proj_b;
};

static inline void gemm(hipStream_t s, const float* A, long lda,
                        const float* W, long ldw, long wks, const float* bias,
                        float* C, long ldc, int M, int N, int K,
                        int tshift, int T, int act, int accum) {
  int tiles_m = (M + 15) / 16;
  int ngrp = (((N + 15) / 16) + 3) / 4;
  long jobs = (long)tiles_m * ngrp;
  dim3 g((unsigned)((jobs + 7) / 8)), b(256);     // 8 waves per WG
  bool kal = (K % 32) == 0;
  if (kal && tshift)
    k_gemm<true,  true ><<<g, b, 0, s>>>(A, lda, W, ldw, wks, bias, C, ldc,
                                         M, N, K, tshift, T, act, accum);
  else if (kal)
    k_gemm<true,  false><<<g, b, 0, s>>>(A, lda, W, ldw, wks, bias, C, ldc,
                                         M, N, K, tshift, T, act, accum);
  else if (tshift)
    k_gemm<false, true ><<<g, b, 0, s>>>(A, lda, W, ldw, wks, bias, C, ldc,
                                         M, N, K, tshift, T, act, accum);
  else
    k_gemm<false, false><<<g, b, 0, s>>>(A, lda, W, ldw, wks, bias, C, ldc,
                                         M, N, K, tshift, T, act, accum);
}

extern "C" void kernel_launch(void* const* d_in, const int* in_sizes, int n_in,
                              void* d_out, int out_size, void* d_ws, size_t ws_size,
                              hipStream_t stream) {
  (void)in_sizes; (void)n_in; (void)out_size; (void)ws_size;
  const int  Bb = 8, T = 2048;
  const long BT = (long)Bb * T;

  auto F = [&](int i) { return (const float*)d_in[i]; };

  // JAX pytree (sorted dict keys) flat order:
  //  0..3  cls_b1, cls_b2, cls_w1, cls_w2
  //  4..   cossm[0..2], enssm[0]   (per block: bn_b,bn_g,conv_w,ln_b,ln_g,
  //        mamba{A_b_log,A_log,D,D_b,conv_b,conv_b_b,conv_w,conv_w_b,
  //              dt_b,dt_b_b,dt_w,dt_w_b,in_proj,out_proj,x_proj,x_proj_b},
  //        [skip_w])
  //  then input_proj, then x
  const float* cls_b1 = F(0);
  const float* cls_b2 = F(1);
  const float* cls_w1 = F(2);
  const float* cls_w2 = F(3);

  BlockP blk[4];
  const int dims_in[4]  = {128, 256, 512, 512};
  const int dims_out[4] = {256, 512, 512, 512};
  int idx = 4;
  for (int i = 0; i < 4; ++i) {
    bool has_skip = (i < 2);
    BlockP& p = blk[i];
    p.bn_b = F(idx + 0); p.bn_g = F(idx + 1); p.conv_w = F(idx + 2);
    p.ln_b = F(idx + 3); p.ln_g = F(idx + 4);
    int m = idx + 5;
    p.A_b_log = F(m + 0);  p.A_log    = F(m + 1);
    p.D       = F(m + 2);  p.D_b      = F(m + 3);
    p.conv_b  = F(m + 4);  p.conv_b_b = F(m + 5);
    p.dconv_w = F(m + 6);  p.dconv_w_b= F(m + 7);
    p.dt_b    = F(m + 8);  p.dt_b_b   = F(m + 9);
    p.dt_w    = F(m + 10); p.dt_w_b   = F(m + 11);
    p.in_proj = F(m + 12); p.out_proj = F(m + 13);
    p.x_proj  = F(m + 14); p.x_proj_b = F(m + 15);
    p.skip_w  = has_skip ? F(m + 16) : nullptr;
    idx += has_skip ? 22 : 21;
  }
  const float* input_proj = F(idx);       // (128, 172)
  const float* x          = F(idx + 1);   // (8, 2048, 172)

  // ---- workspace layout (floats); total ~613 MB ----
  float* ws = (float*)d_ws;
  float* R_h    = ws; ws += BT * 512;
  float* R_hb   = ws; ws += BT * 512;
  float* R_conv = ws; ws += BT * 512;
  float* R_y    = ws; ws += BT * 512;
  float* R_xz   = ws; ws += BT * 2048;
  float* R_xcf  = ws; ws += BT * 1024;
  float* R_xcb  = ws; ws += BT * 1024;
  float* R_dtf  = ws; ws += BT * 1024;
  float* R_dtb  = ws; ws += BT * 1024;
  float* R_dblf = ws; ws += BT * 64;
  float* R_dblb = ws; ws += BT * 64;
  float* R_stat = ws; ws += 2048;         // mean[<=1024] | var at +1024
  float* R_cls  = ws; ws += BT * 1024;

  // ---- input projection: h = x @ input_proj.T   [BT,128], K=172 (edge) ----
  gemm(stream, x, 172, input_proj, 172, 1, nullptr, R_h, 128,
       (int)BT, 128, 172, 0, T, 0, 0);

  for (int i = 0; i < 4; ++i) {
    const BlockP& p = blk[i];
    const int cin = dims_in[i], cout = dims_out[i];
    const int di = 2 * cout, dr = cout / 16, dbs = dr + 32;

    // conv3 "same" as 3 accumulating WMMA GEMMs with time-shifted A rows
    for (int j = 0; j < 3; ++j)
      gemm(stream, R_h, cin, p.conv_w + j, 3L * cin, 3, nullptr, R_conv, cout,
           (int)BT, cout, cin, j - 1, T, 0, j > 0);

    const float* skipsrc = R_h;
    if (p.skip_w) {
      gemm(stream, R_h, cin, p.skip_w, cin, 1, nullptr, R_hb, cout,
           (int)BT, cout, cin, 0, T, 0, 0);
      skipsrc = R_hb;
    }

    // BatchNorm over (B,T) + relu + skip-add  -> R_hb
    k_bnstats<<<cout, 256, 0, stream>>>(R_conv, cout, BT, R_stat, R_stat + 1024);
    {
      long tot = BT * cout;
      k_bn_relu_add<<<(unsigned)((tot + 255) / 256), 256, 0, stream>>>(
          R_conv, R_stat, R_stat + 1024, p.bn_g, p.bn_b, skipsrc, R_hb, cout, tot);
    }

    // Mamba: xz = h @ in_proj.T   [BT, 2*di]
    gemm(stream, R_hb, cout, p.in_proj, cout, 1, nullptr, R_xz, 2 * di,
         (int)BT, 2 * di, cout, 0, T, 0, 0);

    // depthwise causal conv4 + SiLU, both directions
    long tot = BT * di;
    unsigned gr = (unsigned)((tot + 255) / 256);
    k_dwconv_silu<<<gr, 256, 0, stream>>>(R_xz, 2 * di, p.dconv_w, p.conv_b,
                                          R_xcf, T, di, tot, 0);
    k_dwconv_silu<<<gr, 256, 0, stream>>>(R_xz, 2 * di, p.dconv_w_b, p.conv_b_b,
                                          R_xcb, T, di, tot, 1);

    // dbl = xc @ x_proj.T   [BT, dr+32]
    gemm(stream, R_xcf, di, p.x_proj,   di, 1, nullptr, R_dblf, dbs,
         (int)BT, dbs, di, 0, T, 0, 0);
    gemm(stream, R_xcb, di, p.x_proj_b, di, 1, nullptr, R_dblb, dbs,
         (int)BT, dbs, di, 0, T, 0, 0);

    // dt = softplus(dbl[:,:dr] @ dt_w.T + dt_b)   (A = strided column view)
    gemm(stream, R_dblf, dbs, p.dt_w,   dr, 1, p.dt_b,   R_dtf, di,
         (int)BT, di, dr, 0, T, 3, 0);
    gemm(stream, R_dblb, dbs, p.dt_w_b, dr, 1, p.dt_b_b, R_dtb, di,
         (int)BT, di, dr, 0, T, 3, 0);

    // selective scans (y overwrites dt buffer in place)
    int nscan = Bb * di;
    unsigned gs = (unsigned)((nscan + 255) / 256);
    k_scan<<<gs, 256, 0, stream>>>(R_xcf, R_dtf, R_dblf, dbs, dr,
                                   p.A_log,   p.D,   Bb, T, di, 0);
    k_scan<<<gs, 256, 0, stream>>>(R_xcb, R_dtb, R_dblb, dbs, dr,
                                   p.A_b_log, p.D_b, Bb, T, di, 1);

    // gate: (yf + yb) * silu(z) -> reuse R_xcf
    k_gate<<<gr, 256, 0, stream>>>(R_dtf, R_dtb, R_xz, 2 * di, di, R_xcf, tot);

    // out_proj -> R_y
    gemm(stream, R_xcf, di, p.out_proj, di, 1, nullptr, R_y, cout,
         (int)BT, cout, di, 0, T, 0, 0);

    // h = hb + LN(y)  -> R_h (next block input)
    k_ln_add<<<(unsigned)((BT + 7) / 8), 256, 0, stream>>>(
        R_y, p.ln_g, p.ln_b, R_hb, R_h, cout, BT);
  }

  // ---- classifier ----
  gemm(stream, R_h, 512, cls_w1, 512, 1, cls_b1, R_cls, 1024,
       (int)BT, 1024, 512, 0, T, 1, 0);
  gemm(stream, R_cls, 1024, cls_w2, 1024, 1, cls_b2, (float*)d_out, 1002,
       (int)BT, 1002, 1024, 0, T, 0, 0);
}
